// RingDilatedAttentionHybridHilbert_46153718563051
// MI455X (gfx1250) — matrix operand order repaired and checked by reading, not compiled, not run on hardware
//
#include <hip/hip_runtime.h>
#include <stdint.h>

typedef __attribute__((ext_vector_type(16))) _Float16 v16h;
typedef __attribute__((ext_vector_type(8)))  float    v8f;

#define B_  2
#define N_  4096
#define H_  16
#define D_  64

__device__ __forceinline__ uint32_t pack2h(float a, float b){
  union { _Float16 h; unsigned short s; } ua, ub;
  ua.h = (_Float16)a; ub.h = (_Float16)b;
  return (uint32_t)ua.s | ((uint32_t)ub.s << 16);
}

__device__ __forceinline__ float fast_exp2(float x){
#if __has_builtin(__builtin_amdgcn_exp2f)
  return __builtin_amdgcn_exp2f(x);
#else
  return exp2f(x);
#endif
}

union AB { v16h v; uint32_t u[8]; };

// ---------------------------------------------------------------------------
// Zero output (zeros for even positions of heads 8..15) + zero map.
// ---------------------------------------------------------------------------
__global__ void zero_init_kernel(float* __restrict__ out, int* __restrict__ map, int n4){
  int i = blockIdx.x * blockDim.x + threadIdx.x;
  if (i < n4) ((float4*)out)[i] = make_float4(0.f, 0.f, 0.f, 0.f);
  if (i < N_) map[i] = 0;
}

// ---------------------------------------------------------------------------
// Hilbert mapping (exact reference semantics, last-i-wins via atomicMax).
// ---------------------------------------------------------------------------
__global__ void hilbert_map_kernel(int* __restrict__ map){
  int i = blockIdx.x * blockDim.x + threadIdx.x;
  int x = 0, y = 0, d = i;
  #pragma unroll
  for (int s = 1; s < 64; s <<= 1){
    int rx = (d >> 1) & 1;
    int ry = (d ^ rx) & 1;
    if (ry == 0){
      if (rx == 1){ x = 63 - x; y = 63 - y; }
      int t = x; x = y; y = t;
    }
    x += s * rx;
    y += s * ry;
    d >>= 2;
  }
  atomicMax(&map[y * 64 + x], i);
}

// ---------------------------------------------------------------------------
// Pre-pack: Hilbert-gather K/V into f16 streams per (b, head-in-group).
//   kh: [stream][key][32 dwords]        (f16 pairs over d, row-major)
//   vh: [stream][64 d][streamLen/2 dw]  (f16 pairs over KEY, i.e. V^T)
// ---------------------------------------------------------------------------
__global__ __launch_bounds__(128)
void pack_kv_kernel(const float* __restrict__ k, const float* __restrict__ v,
                    const int* __restrict__ map,
                    uint32_t* __restrict__ kh, uint32_t* __restrict__ vh, int group)
{
  __shared__ __align__(16) uint32_t lds[32 * 32];   // V tile, f16 pairs over d
  const int tid = threadIdx.x;
  const int streamLen = group ? 2048 : 4096;
  const int tiles = streamLen >> 5;
  const int tile = blockIdx.x % tiles;
  const int pb   = blockIdx.x / tiles;              // stream = b*8 + hh
  const int b = pb >> 3, hh = pb & 7;
  const int h = group ? (8 + hh) : hh;

  for (int i = tid; i < 1024; i += 128){
    int kk = i >> 5, dp = i & 31;
    int j = (tile << 5) + kk;
    int n = map[group ? ((j << 1) | 1) : j];
    const float* ks = k + (size_t)(((b * N_ + n) * H_ + h) * D_ + (dp << 1));
    const float* vs = v + (size_t)(((b * N_ + n) * H_ + h) * D_ + (dp << 1));
    kh[((size_t)pb * streamLen + j) * 32 + dp] = pack2h(ks[0], ks[1]);
    lds[(kk << 5) + dp] = pack2h(vs[0], vs[1]);
  }
  __syncthreads();
  // transpose write: output dword (d, keypair kp) = {V[2kp][d], V[2kp+1][d]}
  for (int i = tid; i < 1024; i += 128){
    int d = i >> 4, kp = i & 15;
    uint32_t lo = lds[(((kp << 1)    ) << 5) + (d >> 1)];
    uint32_t hi = lds[(((kp << 1) | 1) << 5) + (d >> 1)];
    uint32_t a = (d & 1) ? (lo >> 16) : (lo & 0xffffu);
    uint32_t c = (d & 1) ? (hi >> 16) : (hi & 0xffffu);
    vh[((size_t)pb * 64 + d) * (streamLen >> 1) + (tile << 4) + kp] = a | (c << 16);
  }
}

// ---------------------------------------------------------------------------
// Flash attention on f16 streams. Block = 128 threads = 4 waves.
// Each wave owns 32 q rows (2 A tiles); 64-key tiles; per iteration per wave:
//   16 S-WMMA + 16 PV-WMMA + 4 rowsum-WMMA = 36 WMMAs per 64 keys.
// K/V tiles staged via global_load_async_to_lds_b128 (ASYNCcnt).
// ---------------------------------------------------------------------------
__global__ __launch_bounds__(128)
void attn_kernel(const float* __restrict__ q,
                 const uint32_t* __restrict__ kh, const uint32_t* __restrict__ vh,
                 float* __restrict__ out, int group)
{
  __shared__ __align__(16) uint32_t smQ[128 * 32];   // [128 rows][32 dw] f16 pairs over d
  __shared__ __align__(16) uint32_t smK[64 * 32];    // [64 keys][32 dw]  f16 pairs over d
  __shared__ __align__(16) uint32_t smV[64 * 32];    // [64 d][32 dw]     f16 pairs over key
  __shared__ __align__(16) uint32_t smP[4 * 2 * 512]; // per wave, per q-subtile: [16 rows][32 dw]

  const int tid  = threadIdx.x;
  const int w    = tid >> 5;
  const int lane = tid & 31;
  const int hi   = lane >> 4;
  const int ln   = lane & 15;

  int b, seg, hh, qtile, nKloop, kstart, streamLen;
  if (group == 0){
    int p = blockIdx.x >> 3; qtile = blockIdx.x & 7;       // 64 problems x 8 tiles
    b = p >> 5; int rem = p & 31; seg = rem >> 3; hh = rem & 7;
    nKloop = 1024; kstart = seg * 1024; streamLen = 4096;
  } else {
    int p = blockIdx.x >> 4; qtile = blockIdx.x & 15;      // 16 problems x 16 tiles
    b = p >> 3; hh = p & 7; seg = 0;
    nKloop = 2048; kstart = 0; streamLen = 2048;
  }
  const int h  = group ? (8 + hh) : hh;
  const int pb = b * 8 + hh;
  const int qrow0 = qtile * 128;

  // ---- stage Q as f16, scaled by 0.125 * log2(e) (exp2-domain softmax) ----
  const float qscale = 0.125f * 1.44269504088896340736f;
  for (int i = tid; i < 128 * 32; i += 128){
    int r = i >> 5; int d = (i & 31) << 1;
    int qr = qrow0 + r;
    int n  = (group == 0) ? (seg * 1024 + qr) : ((qr << 1) | 1);
    const float* src = q + (size_t)(((b * N_ + n) * H_ + h) * D_ + d);
    smQ[i] = pack2h(src[0] * qscale, src[1] * qscale);
  }
  __syncthreads();

  // ---- per-wave Q in A-layout: 2 q-subtiles x 2 D halves ----
  AB aq[2][2];
  #pragma unroll
  for (int qt = 0; qt < 2; ++qt)
    #pragma unroll
    for (int j = 0; j < 8; ++j){
      int kp = ((j & 3) << 1) + (hi << 3) + ((j >> 2) << 4);
      int row = w * 32 + qt * 16 + ln;
      aq[qt][0].u[j] = smQ[row * 32 + (kp >> 1)];
      aq[qt][1].u[j] = smQ[row * 32 + (kp >> 1) + 16];
    }

  // constant all-ones B operand (row sums via WMMA)
  AB bOne;
  #pragma unroll
  for (int j = 0; j < 8; ++j) bOne.u[j] = 0x3C003C00u;

  v8f acc[2][4]; v8f accS[2]; float mi[2][8];
  #pragma unroll
  for (int qt = 0; qt < 2; ++qt){
    accS[qt] = (v8f){};
    #pragma unroll
    for (int t = 0; t < 4; ++t) acc[qt][t] = (v8f){};
    #pragma unroll
    for (int r = 0; r < 8; ++r) mi[qt][r] = -3.0e38f;
  }
  const v8f zc = {};
  const int pbase = w * 1024;

  for (int kb = 0; kb < nKloop; kb += 64){
    __syncthreads();
    // ---- async-stage K tile (8KB) and V^T tile (8KB) straight into LDS ----
    {
      const char* ksrc = (const char*)(kh + ((size_t)pb * streamLen + kstart + kb) * 32);
      uint32_t kdst = (uint32_t)(size_t)(&smK[0]);
      #pragma unroll
      for (int it = 0; it < 4; ++it){
        int i = tid + it * 128;                       // 512 x 16B
        uint32_t lo = kdst + i * 16;
        uint64_t ga = (uint64_t)(size_t)(ksrc + i * 16);
        asm volatile("global_load_async_to_lds_b128 %0, %1, off"
                     :: "v"(lo), "v"(ga) : "memory");
      }
      #pragma unroll
      for (int it = 0; it < 4; ++it){
      int i = tid + it * 128;
      int d = i >> 3, c = i & 7;                      // 64 rows x 8 x 16B
      const char* vsrc = (const char*)(vh + ((size_t)pb * 64 + d) * (streamLen >> 1)
                                          + ((kstart + kb) >> 1));
      uint32_t vdst = (uint32_t)(size_t)(&smV[0]) + (uint32_t)(d * 32 + c * 4) * 4u;
      uint64_t ga = (uint64_t)(size_t)(vsrc + c * 16);
      asm volatile("global_load_async_to_lds_b128 %0, %1, off"
                   :: "v"(vdst), "v"(ga) : "memory");
      }
    }
    asm volatile("s_wait_asynccnt 0" ::: "memory");
    __syncthreads();

    // ---- S = Q K^T over 4 key column tiles ----
    v8f s[2][4];
    #pragma unroll
    for (int kt = 0; kt < 4; ++kt){
      AB bk0, bk1;
      #pragma unroll
      for (int j = 0; j < 8; ++j){
        int kp = ((j & 3) << 1) + (hi << 3) + ((j >> 2) << 4);
        int dw = kp >> 1;
        bk0.u[j] = smK[(kt * 16 + ln) * 32 + dw];
        bk1.u[j] = smK[(kt * 16 + ln) * 32 + dw + 16];
      }
      #pragma unroll
      for (int qt = 0; qt < 2; ++qt){
        v8f t = __builtin_amdgcn_wmma_f32_16x16x32_f16(false, aq[qt][0].v, false, bk0.v, (short)0, zc, false, false);
        s[qt][kt] = __builtin_amdgcn_wmma_f32_16x16x32_f16(false, aq[qt][1].v, false, bk1.v, (short)0, t, false, false);
      }
    }

    // ---- online softmax (exp2 domain) over 64 keys; P -> per-wave LDS ----
    #pragma unroll
    for (int qt = 0; qt < 2; ++qt){
      #pragma unroll
      for (int r = 0; r < 8; ++r){
        float red = fmaxf(fmaxf(s[qt][0][r], s[qt][1][r]), fmaxf(s[qt][2][r], s[qt][3][r]));
        red = fmaxf(red, __shfl_xor(red, 1));
        red = fmaxf(red, __shfl_xor(red, 2));
        red = fmaxf(red, __shfl_xor(red, 4));
        red = fmaxf(red, __shfl_xor(red, 8));
        float newm  = fmaxf(mi[qt][r], red);
        float alpha = fast_exp2(mi[qt][r] - newm);
        mi[qt][r] = newm;
        float p0 = fast_exp2(s[qt][0][r] - newm);
        float p1 = fast_exp2(s[qt][1][r] - newm);
        float p2 = fast_exp2(s[qt][2][r] - newm);
        float p3 = fast_exp2(s[qt][3][r] - newm);
        acc[qt][0][r] *= alpha; acc[qt][1][r] *= alpha;
        acc[qt][2][r] *= alpha; acc[qt][3][r] *= alpha;
        accS[qt][r]   *= alpha;
        // pair adjacent columns via neighbor shuffle, store dwords
        float o0 = __shfl_xor(p0, 1);
        float o1 = __shfl_xor(p1, 1);
        float o2 = __shfl_xor(p2, 1);
        float o3 = __shfl_xor(p3, 1);
        int row = r + (hi << 3);
        int base = pbase + qt * 512 + row * 32;
        if ((lane & 1) == 0){
          smP[base      + (ln >> 1)] = pack2h(p0, o0);   // keys  0..15
          smP[base + 16 + (ln >> 1)] = pack2h(p2, o2);   // keys 32..47
        } else {
          smP[base +  8 + (ln >> 1)] = pack2h(o1, p1);   // keys 16..31
          smP[base + 24 + (ln >> 1)] = pack2h(o3, p3);   // keys 48..63
        }
      }
    }
    asm volatile("" ::: "memory");   // keep LDS store->load order (same-wave DS is in-order)

    // ---- P in A-layout (2 K-halves per q-subtile) ----
    AB ap[2][2];
    #pragma unroll
    for (int qt = 0; qt < 2; ++qt)
      #pragma unroll
      for (int j = 0; j < 8; ++j){
        int kp = ((j & 3) << 1) + (hi << 3) + ((j >> 2) << 4);
        ap[qt][0].u[j] = smP[pbase + qt * 512 + ln * 32 + (kp >> 1)];
        ap[qt][1].u[j] = smP[pbase + qt * 512 + ln * 32 + (kp >> 1) + 16];
      }
    // ---- O += P V over 4 d-column tiles; rowsum += P 1 ----
    #pragma unroll
    for (int t = 0; t < 4; ++t){
      AB bv0, bv1;
      #pragma unroll
      for (int j = 0; j < 8; ++j){
        int kp = ((j & 3) << 1) + (hi << 3) + ((j >> 2) << 4);
        bv0.u[j] = smV[(t * 16 + ln) * 32 + (kp >> 1)];
        bv1.u[j] = smV[(t * 16 + ln) * 32 + (kp >> 1) + 16];
      }
      #pragma unroll
      for (int qt = 0; qt < 2; ++qt){
        acc[qt][t] = __builtin_amdgcn_wmma_f32_16x16x32_f16(false, ap[qt][0].v, false, bv0.v, (short)0, acc[qt][t], false, false);
        acc[qt][t] = __builtin_amdgcn_wmma_f32_16x16x32_f16(false, ap[qt][1].v, false, bv1.v, (short)0, acc[qt][t], false, false);
      }
    }
    #pragma unroll
    for (int qt = 0; qt < 2; ++qt){
      accS[qt] = __builtin_amdgcn_wmma_f32_16x16x32_f16(false, ap[qt][0].v, false, bOne.v, (short)0, accS[qt], false, false);
      accS[qt] = __builtin_amdgcn_wmma_f32_16x16x32_f16(false, ap[qt][1].v, false, bOne.v, (short)0, accS[qt], false, false);
    }
  }

  // ---- epilogue: normalize by rowsum accumulator, store fp32 ----
  #pragma unroll
  for (int qt = 0; qt < 2; ++qt)
    #pragma unroll
    for (int r = 0; r < 8; ++r){
      float inv = 1.0f / accS[qt][r];
      int row = r + (hi << 3);
      int qr  = qrow0 + w * 32 + qt * 16 + row;
      int n   = (group == 0) ? (seg * 1024 + qr) : ((qr << 1) | 1);
      float* dst = out + (size_t)(((b * N_ + n) * H_ + h) * D_);
      dst[ 0 + ln] = acc[qt][0][r] * inv;
      dst[16 + ln] = acc[qt][1][r] * inv;
      dst[32 + ln] = acc[qt][2][r] * inv;
      dst[48 + ln] = acc[qt][3][r] * inv;
    }
}

// ---------------------------------------------------------------------------
extern "C" void kernel_launch(void* const* d_in, const int* in_sizes, int n_in,
                              void* d_out, int out_size, void* d_ws, size_t ws_size,
                              hipStream_t stream)
{
  const float* q = (const float*)d_in[0];
  const float* k = (const float*)d_in[1];
  const float* v = (const float*)d_in[2];
  float* out = (float*)d_out;

  // workspace layout: map (16KB) | kh0 8MB | vh0 8MB | kh1 4MB | vh1 4MB
  int* map = (int*)d_ws;
  uint32_t* kh0 = (uint32_t*)((char*)d_ws + 4096 * sizeof(int));
  uint32_t* vh0 = kh0 + (size_t)16 * 4096 * 32;
  uint32_t* kh1 = vh0 + (size_t)16 * 64 * 2048;
  uint32_t* vh1 = kh1 + (size_t)16 * 2048 * 32;

  const int n4 = (B_ * N_ * H_ * D_) / 4;
  zero_init_kernel<<<(n4 + 255) / 256, 256, 0, stream>>>(out, map, n4);
  hilbert_map_kernel<<<N_ / 256, 256, 0, stream>>>(map);

  pack_kv_kernel<<<16 * (4096 / 32), 128, 0, stream>>>(k, v, map, kh0, vh0, 0);
  pack_kv_kernel<<<16 * (2048 / 32), 128, 0, stream>>>(k, v, map, kh1, vh1, 1);

  // group 0: 64 problems x 8 q-tiles of 128 rows
  attn_kernel<<<512, 128, 0, stream>>>(q, kh0, vh0, out, 0);
  // group 1: 16 problems x 16 q-tiles of 128 rows
  attn_kernel<<<256, 128, 0, stream>>>(q, kh1, vh1, out, 1);
}
